// Attention_27779848470884
// MI455X (gfx1250) — compile-verified, matrix operand order checked by reading, not run on hardware
//
#include <hip/hip_runtime.h>
#include <hip/hip_fp16.h>

// ---------------------------------------------------------------------------
// Attention prefill for MI455X (gfx1250, wave32, WMMA 16x16x32 f16).
// B=4, S=2048, D=2048, H=16, HD=128, pos=0 (prefill).
// Compute-bound (~412 GFLOP vs ~15us of HBM traffic at 23.3 TB/s): all
// matmuls run through v_wmma_f32_16x16x32_f16 with fp32 accumulation.
// Data movement uses the CDNA5 paths: GLOBAL_LOAD_ASYNC_TO_LDS_B128 for
// double-buffered GEMM staging (ASYNCcnt) and the Tensor Data Mover for the
// flash-attention K tiles (TENSORcnt), with guarded fallbacks.
// ---------------------------------------------------------------------------

#define BATCH   4
#define SEQ     2048
#define DMODEL  2048
#define NHEAD   16
#define HEADD   128
#define EMB     (NHEAD * HEADD)   // 2048
#define MROWS   (BATCH * SEQ)     // 8192

typedef __attribute__((ext_vector_type(16))) _Float16 v16h;
typedef __attribute__((ext_vector_type(8)))  _Float16 v8h;
typedef __attribute__((ext_vector_type(8)))  float    v8f;
typedef unsigned int u32x4 __attribute__((ext_vector_type(4)));
typedef int          i32x8 __attribute__((ext_vector_type(8)));
typedef int          i32x4 __attribute__((ext_vector_type(4)));
typedef int          v4i   __attribute__((vector_size(16)));   // async-LDS payload

#if defined(__gfx1250__) && __has_builtin(__builtin_amdgcn_global_load_async_to_lds_b128)
#define HAVE_ASYNC_LDS 1
#else
#define HAVE_ASYNC_LDS 0
#endif

#if defined(__gfx1250__) && __has_builtin(__builtin_amdgcn_tensor_load_to_lds) && \
    __has_builtin(__builtin_amdgcn_s_wait_tensorcnt)
#define HAVE_TDM 1
#else
#define HAVE_TDM 0
#endif

// Address-space casts via integer round-trip (flat LDS pointers carry the LDS
// byte offset in their low 32 bits per CDNA5 aperture rules). The async-LDS
// builtins take typed b128 pointers: v4i addrspace(1)* / v4i addrspace(3)*.
__device__ __forceinline__ __attribute__((address_space(1))) v4i*
gcast(const void* p) {
  return (__attribute__((address_space(1))) v4i*)(unsigned long long)(size_t)p;
}
__device__ __forceinline__ __attribute__((address_space(3))) v4i*
lcast(const void* p) {
  return (__attribute__((address_space(3))) v4i*)(unsigned)(size_t)p;
}

__device__ __forceinline__ void wait_async0() {
#if HAVE_ASYNC_LDS
#if __has_builtin(__builtin_amdgcn_s_wait_asynccnt)
  __builtin_amdgcn_s_wait_asynccnt(0);
#else
  asm volatile("s_wait_asynccnt 0x0" ::: "memory");
#endif
#endif
}

// One 2D tile (rows x rowElems of 2-byte elements) global -> LDS via the TDM.
// D# packed per CDNA5 ISA 8.3/8.4 (group0: ctrl/lds/global/type, group1:
// data_size + dims/strides). Issued once per wave; tracked with TENSORcnt.
__device__ __forceinline__ void tdm_load_2d_f16(const void* gsrc, void* ldsDst,
                                                unsigned rows, unsigned rowElems,
                                                unsigned strideElems,
                                                unsigned tensorRows) {
#if HAVE_TDM
  const unsigned long long ga = (unsigned long long)(size_t)gsrc;
  const unsigned lds = (unsigned)(size_t)ldsDst;
  u32x4 g0;
  g0[0] = 1u;                                            // count=1, user D#
  g0[1] = lds;                                           // lds_addr
  g0[2] = (unsigned)(ga & 0xffffffffull);                // global_addr[31:0]
  g0[3] = (unsigned)((ga >> 32) & 0x01ffffffull) | (2u << 30);  // [56:32]|type=2
  i32x8 g1;
  g1[0] = (int)(1u << 16);                               // data_size=1 (2B)
  g1[1] = (int)((rowElems & 0xffffu) << 16);             // tensor_dim0 lo
  g1[2] = (int)((rowElems >> 16) | ((tensorRows & 0xffffu) << 16)); // dim0 hi|dim1 lo
  g1[3] = (int)((tensorRows >> 16) | ((rowElems & 0xffffu) << 16)); // dim1 hi|tile_dim0
  g1[4] = (int)(rows & 0xffffu);                         // tile_dim1 (tile_dim2=0)
  g1[5] = (int)strideElems;                              // tensor_dim0_stride lo
  g1[6] = 0;
  g1[7] = 0;
  i32x4 gz = {0, 0, 0, 0};
#if defined(__clang_major__) && (__clang_major__ >= 23)
  i32x8 gz8 = {0, 0, 0, 0, 0, 0, 0, 0};
  __builtin_amdgcn_tensor_load_to_lds(g0, g1, gz, gz, gz8, 0);
#else
  __builtin_amdgcn_tensor_load_to_lds(g0, g1, gz, gz, 0);
#endif
#else
  (void)gsrc; (void)ldsDst; (void)rows; (void)rowElems;
  (void)strideElems; (void)tensorRows;
#endif
}

// Load a 16x32 f16 fragment (A-operand layout per CDNA5 ISA 7.12.2) from a
// row-major tile with leading dimension ld (halves). The same layout serves
// as the B operand of D = A * B^T when B is stored row-major [N][K].
__device__ __forceinline__ v16h load_frag16(const _Float16* tile, int ld, int lane) {
  const int r  = lane & 15;            // row (M for A, N for B)
  const int kq = (lane >> 4) << 3;     // lanes 16-31 hold K=8..15 / 24..31
  const _Float16* p = tile + r * ld + kq;
  v8h lo = *(const v8h*)(p);           // K = kq .. kq+7
  v8h hi = *(const v8h*)(p + 16);      // K = 16+kq .. 16+kq+7
  v16h f;
#pragma unroll
  for (int i = 0; i < 8; ++i) { f[i] = lo[i]; f[i + 8] = hi[i]; }
  return f;
}

__device__ __forceinline__ v8f wmma_f16(v16h a, v16h b, v8f c) {
  return __builtin_amdgcn_wmma_f32_16x16x32_f16(
      false, a, false, b, (short)0, c, false, false);
}

// ---------------------------------------------------------------------------
// Kernel 1: fp32 -> f16 conversion (grid-stride)
// ---------------------------------------------------------------------------
__global__ __launch_bounds__(256) void f32_to_f16_kernel(
    const float* __restrict__ src, _Float16* __restrict__ dst, size_t n) {
  size_t i = (size_t)blockIdx.x * blockDim.x + threadIdx.x;
  const size_t stride = (size_t)gridDim.x * blockDim.x;
  for (; i < n; i += stride) dst[i] = (_Float16)src[i];
}

// ---------------------------------------------------------------------------
// Kernel 2: fused QKV projection + RoPE + scatter into [B,H,S,HD] f16.
// C = A * W^T. Block: 256 threads = 8 waves (4x2), block tile 128x64, K=32
// steps, async double-buffered LDS staging. blockIdx.z selects Q/K/V; RoPE
// applied to Q/K in the epilogue via lane-pair shuffles.
// ---------------------------------------------------------------------------
__global__ __launch_bounds__(256) void qkv_gemm_rope_kernel(
    const _Float16* __restrict__ A,
    const _Float16* __restrict__ Wq16,
    const _Float16* __restrict__ Wk16,
    const _Float16* __restrict__ Wv16,
    _Float16* __restrict__ Qo,
    _Float16* __restrict__ Ko,
    _Float16* __restrict__ Vo,
    const float* __restrict__ fcos,
    const float* __restrict__ fsin,
    const int* __restrict__ posp) {
  __shared__ _Float16 As[2][128 * 32];
  __shared__ _Float16 Bs[2][64 * 32];

  const int z = blockIdx.z;
  const _Float16* Bw = (z == 0) ? Wq16 : (z == 1) ? Wk16 : Wv16;
  _Float16* Out      = (z == 0) ? Qo   : (z == 1) ? Ko   : Vo;
  const bool doRope  = (z != 2);

  const int t    = threadIdx.x;
  const int lane = t & 31;
  const int w    = t >> 5;
  const int wm   = w >> 1;       // 0..3
  const int wn   = w & 1;        // 0..1
  const int m0   = blockIdx.y * 128;
  const int n0   = blockIdx.x * 64;

  v8f acc[2][2];
#pragma unroll
  for (int i = 0; i < 2; ++i)
#pragma unroll
    for (int j = 0; j < 2; ++j)
#pragma unroll
      for (int r = 0; r < 8; ++r) acc[i][j][r] = 0.0f;

  const int ar = t >> 1, ac = (t & 1) << 4;   // A stage: 16 halves/thread
  const int br = t >> 2, bc = (t & 3) << 3;   // B stage:  8 halves/thread
  const _Float16* Arow = A  + (size_t)(m0 + ar) * DMODEL + ac;
  const _Float16* Brow = Bw + (size_t)(n0 + br) * DMODEL + bc;

  auto stage = [&](int buf, int k0) {
#if HAVE_ASYNC_LDS
    __builtin_amdgcn_global_load_async_to_lds_b128(
        gcast(Arow + k0),     lcast(&As[buf][ar * 32 + ac]), 0, 0);
    __builtin_amdgcn_global_load_async_to_lds_b128(
        gcast(Arow + k0 + 8), lcast(&As[buf][ar * 32 + ac + 8]), 0, 0);
    __builtin_amdgcn_global_load_async_to_lds_b128(
        gcast(Brow + k0),     lcast(&Bs[buf][br * 32 + bc]), 0, 0);
#else
    *(v8h*)(&As[buf][ar * 32 + ac])     = *(const v8h*)(Arow + k0);
    *(v8h*)(&As[buf][ar * 32 + ac + 8]) = *(const v8h*)(Arow + k0 + 8);
    *(v8h*)(&Bs[buf][br * 32 + bc])     = *(const v8h*)(Brow + k0);
#endif
  };

  stage(0, 0);
  int buf = 0;
  for (int k0 = 0; k0 < DMODEL; k0 += 32) {
    wait_async0();
    __syncthreads();
    if (k0 + 32 < DMODEL) {
      stage(buf ^ 1, k0 + 32);
      __builtin_prefetch(Arow + k0 + 64, 0, 1);   // global_prefetch_b8
      __builtin_prefetch(Brow + k0 + 64, 0, 1);
    }
    const _Float16* At = &As[buf][(wm * 32) * 32];
    const _Float16* Bt = &Bs[buf][(wn * 32) * 32];
    v16h a0 = load_frag16(At,           32, lane);
    v16h a1 = load_frag16(At + 16 * 32, 32, lane);
    v16h b0 = load_frag16(Bt,           32, lane);
    v16h b1 = load_frag16(Bt + 16 * 32, 32, lane);
    acc[0][0] = wmma_f16(a0, b0, acc[0][0]);
    acc[0][1] = wmma_f16(a0, b1, acc[0][1]);
    acc[1][0] = wmma_f16(a1, b0, acc[1][0]);
    acc[1][1] = wmma_f16(a1, b1, acc[1][1]);
    buf ^= 1;
  }

  const int pos    = *posp;
  const int lane15 = lane & 15;
  const int half8  = (lane >> 4) << 3;   // C frag: lanes 16-31 hold rows M+8
#pragma unroll
  for (int fm = 0; fm < 2; ++fm) {
#pragma unroll
    for (int fn = 0; fn < 2; ++fn) {
      const int ncol = n0 + wn * 32 + fn * 16 + lane15;   // e in [0, EMB)
      const int hh   = ncol >> 7;
      const int dd   = ncol & (HEADD - 1);
#pragma unroll
      for (int r = 0; r < 8; ++r) {
        const int mrow = m0 + wm * 32 + fm * 16 + half8 + r;
        const int ss   = mrow & (SEQ - 1);
        float v = acc[fm][fn][r];
        if (doRope) {
          const int jj = dd >> 1;
          const float cv = fcos[(size_t)(pos + ss) * (HEADD / 2) + jj];
          const float sv = fsin[(size_t)(pos + ss) * (HEADD / 2) + jj];
          const float partner = __shfl_xor(v, 1, 32);
          v = (lane15 & 1) ? (partner * sv + v * cv)    // odd  e: x0*sin + x1*cos
                           : (v * cv - partner * sv);   // even e: x0*cos - x1*sin
        }
        const int bb = mrow >> 11;
        Out[(((size_t)bb * NHEAD + hh) * SEQ + ss) * HEADD + dd] = (_Float16)v;
      }
    }
  }
}

// ---------------------------------------------------------------------------
// Kernel 3: causal flash attention, prefill (pos==0 path).
// Grid: (SEQ/128, B*H). Block 256 = 8 waves, wave w owns 16 q-rows.
// K tile (32x128 f16) arrives via the Tensor Data Mover (wave 0 issues one
// tensor_load_to_lds, waits TENSORcnt); V is staged transposed by VALU so the
// P*V B-fragment reads contiguous K. Per tile: 8 wmma (S) + 8 wmma (O).
// ---------------------------------------------------------------------------
__global__ __launch_bounds__(256) void flash_attn_kernel(
    const _Float16* __restrict__ Qf,   // [B,H,S,HD]
    const _Float16* __restrict__ Kf,   // [B,H,S,HD]
    const _Float16* __restrict__ Vf,   // [B,H,S,HD]
    _Float16* __restrict__ Of) {       // [B,S,EMB]
  __shared__ _Float16 Ks[32 * 128];        // K tile, row-major [key][d]
  __shared__ _Float16 Vt[128 * 40];        // V tile transposed [d][key], ld=40
  __shared__ _Float16 Pw[8 * 16 * 32];     // per-wave P scratch [16 q][32 key]

  const int t    = threadIdx.x;
  const int lane = t & 31;
  const int w    = t >> 5;
  const int bh   = blockIdx.y;
  const int bb   = bh >> 4;
  const int hh   = bh & (NHEAD - 1);
  const int q0   = blockIdx.x * 128;
  const int qw   = q0 + w * 16;

  const _Float16* Qbase = Qf + (size_t)bh * SEQ * HEADD;
  const _Float16* Kbase = Kf + (size_t)bh * SEQ * HEADD;
  const _Float16* Vbase = Vf + (size_t)bh * SEQ * HEADD;

  v16h aq[4];
#pragma unroll
  for (int kc = 0; kc < 4; ++kc)
    aq[kc] = load_frag16(Qbase + (size_t)qw * HEADD + kc * 32, HEADD, lane);

  v8f o[8];
#pragma unroll
  for (int j = 0; j < 8; ++j)
#pragma unroll
    for (int r = 0; r < 8; ++r) o[j][r] = 0.0f;

  float mrun[8], lrun[8];
#pragma unroll
  for (int r = 0; r < 8; ++r) { mrun[r] = -1e30f; lrun[r] = 0.0f; }

  const int lane15 = lane & 15;
  const int half8  = (lane >> 4) << 3;
  const float sc = 0.08838834764831845f;   // 1/sqrt(HD)
  _Float16* Pme = Pw + w * (16 * 32);

  const int sr = t >> 3, scol = (t & 7) << 4;   // stage: row 0..31, 16 halves
  const int kend = q0 + 128;                    // causal bound for this block

  for (int k0 = 0; k0 < kend; k0 += 32) {
    __syncthreads();   // previous iteration's Ks/Vt consumers are done
#if HAVE_TDM
    if (w == 0)
      tdm_load_2d_f16(Kbase + (size_t)k0 * HEADD, Ks,
                      /*rows=*/32, /*rowElems=*/HEADD,
                      /*strideElems=*/HEADD, /*tensorRows=*/SEQ);
#elif HAVE_ASYNC_LDS
    __builtin_amdgcn_global_load_async_to_lds_b128(
        gcast(Kbase + (size_t)(k0 + sr) * HEADD + scol),
        lcast(Ks + sr * 128 + scol), 0, 0);
    __builtin_amdgcn_global_load_async_to_lds_b128(
        gcast(Kbase + (size_t)(k0 + sr) * HEADD + scol + 8),
        lcast(Ks + sr * 128 + scol + 8), 0, 0);
#else
    *(v8h*)(Ks + sr * 128 + scol)     = *(const v8h*)(Kbase + (size_t)(k0 + sr) * HEADD + scol);
    *(v8h*)(Ks + sr * 128 + scol + 8) = *(const v8h*)(Kbase + (size_t)(k0 + sr) * HEADD + scol + 8);
#endif
    {   // V transpose staging (TDM/async cannot transpose)
      v8h va = *(const v8h*)(Vbase + (size_t)(k0 + sr) * HEADD + scol);
      v8h vb = *(const v8h*)(Vbase + (size_t)(k0 + sr) * HEADD + scol + 8);
#pragma unroll
      for (int i = 0; i < 8; ++i) {
        Vt[(scol + i) * 40 + sr]     = va[i];
        Vt[(scol + 8 + i) * 40 + sr] = vb[i];
      }
    }
#if HAVE_TDM
    if (w == 0) __builtin_amdgcn_s_wait_tensorcnt(0);
#elif HAVE_ASYNC_LDS
    wait_async0();
#endif
    __syncthreads();

    // S = Q * K^T : two 16x16 score frags (keys k0..+15, k0+16..+31)
    v8f s0, s1;
#pragma unroll
    for (int r = 0; r < 8; ++r) { s0[r] = 0.0f; s1[r] = 0.0f; }
#pragma unroll
    for (int kc = 0; kc < 4; ++kc) {
      v16h b0 = load_frag16(Ks + kc * 32,            128, lane);
      v16h b1 = load_frag16(Ks + 16 * 128 + kc * 32, 128, lane);
      s0 = wmma_f16(aq[kc], b0, s0);
      s1 = wmma_f16(aq[kc], b1, s1);
    }

    // Online softmax. C frag: lane = key column, vgpr r + half8 = q row.
    const int key0 = k0 + lane15;
    const int key1 = k0 + 16 + lane15;
#pragma unroll
    for (int r = 0; r < 8; ++r) {
      const int qrow = qw + half8 + r;
      float x0 = s0[r] * sc; if (key0 > qrow) x0 = -1e30f;
      float x1 = s1[r] * sc; if (key1 > qrow) x1 = -1e30f;
      float mx = fmaxf(x0, x1);
      mx = fmaxf(mx, __shfl_xor(mx, 1, 32));
      mx = fmaxf(mx, __shfl_xor(mx, 2, 32));
      mx = fmaxf(mx, __shfl_xor(mx, 4, 32));
      mx = fmaxf(mx, __shfl_xor(mx, 8, 32));   // stays within 16-lane row group
      const float mnew  = fmaxf(mrun[r], mx);
      const float rescl = __expf(mrun[r] - mnew);
      const float p0 = __expf(x0 - mnew);
      const float p1 = __expf(x1 - mnew);
      float rs = p0 + p1;
      rs += __shfl_xor(rs, 1, 32);
      rs += __shfl_xor(rs, 2, 32);
      rs += __shfl_xor(rs, 4, 32);
      rs += __shfl_xor(rs, 8, 32);
      lrun[r] = lrun[r] * rescl + rs;
      mrun[r] = mnew;
#pragma unroll
      for (int j = 0; j < 8; ++j) o[j][r] *= rescl;
      Pme[(half8 + r) * 32 + lane15]      = (_Float16)p0;
      Pme[(half8 + r) * 32 + 16 + lane15] = (_Float16)p1;
    }
    __syncthreads();   // expose per-wave P with A-fragment layout

    // O += P * V : P is a 16x32 A-frag, V^T rows give B-frags per d-chunk.
    v16h pf = load_frag16(Pme, 32, lane);
#pragma unroll
    for (int j = 0; j < 8; ++j) {
      v16h bv = load_frag16(Vt + (j * 16) * 40, 40, lane);
      o[j] = wmma_f16(pf, bv, o[j]);
    }
  }

  // Normalize and scatter to [B, S, EMB] f16 for the output projection.
#pragma unroll
  for (int j = 0; j < 8; ++j) {
#pragma unroll
    for (int r = 0; r < 8; ++r) {
      const int qrow = qw + half8 + r;
      const float v = o[j][r] / lrun[r];
      const int dd = j * 16 + lane15;
      Of[((size_t)bb * SEQ + qrow) * EMB + hh * HEADD + dd] = (_Float16)v;
    }
  }
}

// ---------------------------------------------------------------------------
// Kernel 4: output projection, C = A * Wo^T, fp32 out (same async tiling).
// ---------------------------------------------------------------------------
__global__ __launch_bounds__(256) void out_proj_gemm_kernel(
    const _Float16* __restrict__ A,    // [MROWS, EMB]
    const _Float16* __restrict__ W,    // [DMODEL, EMB]
    float* __restrict__ Out) {         // [MROWS, DMODEL] = d_out
  __shared__ _Float16 As[2][128 * 32];
  __shared__ _Float16 Bs[2][64 * 32];

  const int t    = threadIdx.x;
  const int lane = t & 31;
  const int w    = t >> 5;
  const int wm   = w >> 1, wn = w & 1;
  const int m0   = blockIdx.y * 128;
  const int n0   = blockIdx.x * 64;

  v8f acc[2][2];
#pragma unroll
  for (int i = 0; i < 2; ++i)
#pragma unroll
    for (int j = 0; j < 2; ++j)
#pragma unroll
      for (int r = 0; r < 8; ++r) acc[i][j][r] = 0.0f;

  const int ar = t >> 1, ac = (t & 1) << 4;
  const int br = t >> 2, bc = (t & 3) << 3;
  const _Float16* Arow = A + (size_t)(m0 + ar) * EMB + ac;
  const _Float16* Brow = W + (size_t)(n0 + br) * EMB + bc;

  auto stage = [&](int buf, int k0) {
#if HAVE_ASYNC_LDS
    __builtin_amdgcn_global_load_async_to_lds_b128(
        gcast(Arow + k0),     lcast(&As[buf][ar * 32 + ac]), 0, 0);
    __builtin_amdgcn_global_load_async_to_lds_b128(
        gcast(Arow + k0 + 8), lcast(&As[buf][ar * 32 + ac + 8]), 0, 0);
    __builtin_amdgcn_global_load_async_to_lds_b128(
        gcast(Brow + k0),     lcast(&Bs[buf][br * 32 + bc]), 0, 0);
#else
    *(v8h*)(&As[buf][ar * 32 + ac])     = *(const v8h*)(Arow + k0);
    *(v8h*)(&As[buf][ar * 32 + ac + 8]) = *(const v8h*)(Arow + k0 + 8);
    *(v8h*)(&Bs[buf][br * 32 + bc])     = *(const v8h*)(Brow + k0);
#endif
  };

  stage(0, 0);
  int buf = 0;
  for (int k0 = 0; k0 < EMB; k0 += 32) {
    wait_async0();
    __syncthreads();
    if (k0 + 32 < EMB) {
      stage(buf ^ 1, k0 + 32);
      __builtin_prefetch(Arow + k0 + 64, 0, 1);
      __builtin_prefetch(Brow + k0 + 64, 0, 1);
    }
    const _Float16* At = &As[buf][(wm * 32) * 32];
    const _Float16* Bt = &Bs[buf][(wn * 32) * 32];
    v16h a0 = load_frag16(At,           32, lane);
    v16h a1 = load_frag16(At + 16 * 32, 32, lane);
    v16h b0 = load_frag16(Bt,           32, lane);
    v16h b1 = load_frag16(Bt + 16 * 32, 32, lane);
    acc[0][0] = wmma_f16(a0, b0, acc[0][0]);
    acc[0][1] = wmma_f16(a0, b1, acc[0][1]);
    acc[1][0] = wmma_f16(a1, b0, acc[1][0]);
    acc[1][1] = wmma_f16(a1, b1, acc[1][1]);
    buf ^= 1;
  }

  const int lane15 = lane & 15;
  const int half8  = (lane >> 4) << 3;
#pragma unroll
  for (int fm = 0; fm < 2; ++fm) {
#pragma unroll
    for (int fn = 0; fn < 2; ++fn) {
      const int ncol = n0 + wn * 32 + fn * 16 + lane15;
#pragma unroll
      for (int r = 0; r < 8; ++r) {
        const int mrow = m0 + wm * 32 + fm * 16 + half8 + r;
        Out[(size_t)mrow * DMODEL + ncol] = acc[fm][fn][r];
      }
    }
  }
}

// ---------------------------------------------------------------------------
// Host launcher. Inputs: h, Wq, Wk, Wv, Wo, K_cache, V_cache, fcos, fsin, pos.
// Workspace (f16): hA 32MB | W* 4x8MB | Q/K/V [B,H,S,HD] 3x32MB | attn-out
// [B,S,EMB] 32MB  (~192MB total).
// ---------------------------------------------------------------------------
extern "C" void kernel_launch(void* const* d_in, const int* in_sizes, int n_in,
                              void* d_out, int out_size, void* d_ws, size_t ws_size,
                              hipStream_t stream) {
  (void)in_sizes; (void)n_in; (void)out_size; (void)ws_size;

  const float* h    = (const float*)d_in[0];
  const float* Wq   = (const float*)d_in[1];
  const float* Wk   = (const float*)d_in[2];
  const float* Wv   = (const float*)d_in[3];
  const float* Wo   = (const float*)d_in[4];
  const float* fcos = (const float*)d_in[7];
  const float* fsin = (const float*)d_in[8];
  const int*   posp = (const int*)d_in[9];
  float* out = (float*)d_out;

  char* ws = (char*)d_ws;
  const size_t nHA = (size_t)MROWS * DMODEL;
  const size_t nW  = (size_t)EMB * DMODEL;
  const size_t nT  = (size_t)BATCH * NHEAD * SEQ * HEADD;

  size_t off = 0;
  _Float16* hA   = (_Float16*)(ws + off); off += nHA * 2;
  _Float16* Wq16 = (_Float16*)(ws + off); off += nW * 2;
  _Float16* Wk16 = (_Float16*)(ws + off); off += nW * 2;
  _Float16* Wv16 = (_Float16*)(ws + off); off += nW * 2;
  _Float16* Wo16 = (_Float16*)(ws + off); off += nW * 2;
  _Float16* Qf   = (_Float16*)(ws + off); off += nT * 2;
  _Float16* Kf   = (_Float16*)(ws + off); off += nT * 2;
  _Float16* Vf   = (_Float16*)(ws + off); off += nT * 2;
  _Float16* Oat  = (_Float16*)(ws + off); off += nT * 2;

  f32_to_f16_kernel<<<4096, 256, 0, stream>>>(h,  hA,   nHA);
  f32_to_f16_kernel<<<2048, 256, 0, stream>>>(Wq, Wq16, nW);
  f32_to_f16_kernel<<<2048, 256, 0, stream>>>(Wk, Wk16, nW);
  f32_to_f16_kernel<<<2048, 256, 0, stream>>>(Wv, Wv16, nW);
  f32_to_f16_kernel<<<2048, 256, 0, stream>>>(Wo, Wo16, nW);

  dim3 gQKV(EMB / 64, MROWS / 128, 3);
  qkv_gemm_rope_kernel<<<gQKV, 256, 0, stream>>>(hA, Wq16, Wk16, Wv16,
                                                 Qf, Kf, Vf, fcos, fsin, posp);

  dim3 gFA(SEQ / 128, BATCH * NHEAD);
  flash_attn_kernel<<<gFA, 256, 0, stream>>>(Qf, Kf, Vf, Oat);

  dim3 gPR(DMODEL / 64, MROWS / 128);
  out_proj_gemm_kernel<<<gPR, 256, 0, stream>>>(Oat, Wo16, out);
}
